// FractalAttention_71004399337601
// MI455X (gfx1250) — matrix operand order repaired and checked by reading, not compiled, not run on hardware
//
#include <hip/hip_runtime.h>
#include <hip/hip_bf16.h>
#include <cstdint>
#include <cstddef>

#ifndef USE_TDM
#define USE_TDM 1
#endif

namespace {
constexpr int B = 4, S = 4096, D = 64;
constexpr float TAU = 0.1f;
constexpr int WPB = 4; // waves per block in the main kernel (wave32)
}

typedef __attribute__((ext_vector_type(16))) _Float16 v16h;
typedef __attribute__((ext_vector_type(8)))  _Float16 v8h;
typedef __attribute__((ext_vector_type(8)))  float    v8f;
typedef __attribute__((ext_vector_type(4)))  float    v4f;
typedef __attribute__((ext_vector_type(4)))  unsigned int u32x4;
typedef __attribute__((ext_vector_type(8)))  int i32x8;
typedef __attribute__((ext_vector_type(4)))  int i32x4;

static __device__ __forceinline__ float fast_sqrtf(float x) {
#if __has_builtin(__builtin_amdgcn_sqrtf)
  return __builtin_amdgcn_sqrtf(x);   // raw v_sqrt_f32, no denorm-fixup ladder
#else
  return __fsqrt_rn(x);
#endif
}

static __device__ __forceinline__ v8f wmma_f32_f16(v16h a, v16h b, v8f c) {
  // D = A(16x32 f16) * B(32x16 f16) + C(16x16 f32)
  return __builtin_amdgcn_wmma_f32_16x16x32_f16(false, a, false, b, (short)0, c,
                                                false, false);
}

static __device__ __forceinline__ v16h ld16h(const _Float16* p0, const _Float16* p1) {
  v8h lo = *(const v8h*)p0;
  v8h hi = *(const v8h*)p1;
  v16h r;
#pragma unroll
  for (int i = 0; i < 8; ++i) { r[i] = lo[i]; r[8 + i] = hi[i]; }
  return r;
}

#if USE_TDM
// Issue one TDM 2-D tile load (global -> LDS). Per-wave instruction; EXEC ignored.
static __device__ __forceinline__ void tdm_load_2d(
    unsigned lds_off, const void* gptr,
    unsigned tensor_d0, unsigned tensor_d1,
    unsigned tile_d0, unsigned tile_d1,
    unsigned long long stride0_elems)
{
  unsigned long long ga = (unsigned long long)(uintptr_t)gptr;
  u32x4 g0;
  g0[0] = 1u;                                      // count=1, user descriptor
  g0[1] = lds_off;                                 // lds_addr (bytes)
  g0[2] = (unsigned)(ga & 0xFFFFFFFFu);            // global_addr[31:0]
  g0[3] = (unsigned)((ga >> 32) & 0x01FFFFFFu) | (2u << 30); // addr[56:32] | type=2
  i32x8 g1;
  g1[0] = (int)(1u << 16);                         // data_size=1 (2 bytes/elem)
  g1[1] = (int)((tensor_d0 & 0xFFFFu) << 16);      // tensor_dim0[15:0] @ bits[63:48]
  g1[2] = (int)((tensor_d0 >> 16) | ((tensor_d1 & 0xFFFFu) << 16));
  g1[3] = (int)((tensor_d1 >> 16) | (tile_d0 << 16));   // tile_dim0
  g1[4] = (int)tile_d1;                            // tile_dim1 (tile_dim2 = 0)
  g1[5] = (int)(unsigned)(stride0_elems & 0xFFFFFFFFull); // dim0 stride
  g1[6] = (int)(unsigned)((stride0_elems >> 32) & 0xFFFFull);
  g1[7] = 0;
  i32x4 z4 = {0, 0, 0, 0};
#if defined(__clang_major__) && (__clang_major__ >= 23)
  i32x8 z8 = {0, 0, 0, 0, 0, 0, 0, 0};
  __builtin_amdgcn_tensor_load_to_lds(g0, g1, z4, z4, z8, 0);
#else
  __builtin_amdgcn_tensor_load_to_lds(g0, g1, z4, z4, 0);
#endif
}
#endif

// -------- Kernel 1: gather + f16 convert (+ transposed copy) + row norms ------
__global__ __launch_bounds__(256) void FractalAttention_embed(
    const int* __restrict__ X, const float* __restrict__ emb,
    _Float16* __restrict__ xh, _Float16* __restrict__ xhT, float* __restrict__ xn)
{
  const int wave = threadIdx.x >> 5, lane = threadIdx.x & 31;
  const int row = blockIdx.x * 8 + wave;           // row in [0, B*S)
  const int b = row / S, s = row % S;
  const int tok = X[row];
  const float* e = emb + (size_t)tok * D;
  float v0 = e[2 * lane], v1 = e[2 * lane + 1];
  xh[(size_t)row * D + 2 * lane]     = (_Float16)v0;
  xh[(size_t)row * D + 2 * lane + 1] = (_Float16)v1;
  xhT[((size_t)b * D + 2 * lane)     * S + s] = (_Float16)v0;
  xhT[((size_t)b * D + 2 * lane + 1) * S + s] = (_Float16)v1;
  float p = v0 * v0 + v1 * v1;
  p += __shfl_xor(p, 1, 32);
  p += __shfl_xor(p, 2, 32);
  p += __shfl_xor(p, 4, 32);
  p += __shfl_xor(p, 8, 32);
  p += __shfl_xor(p, 16, 32);
  if (lane == 0) xn[row] = p;
}

// -------- Kernel 2: fused distance-kernel attention --------------------------
// PASS1: compute softmax denominators (rowsum of exp(w)), write 1/rowsum.
// PASS2: recompute e, write normalized attn, accumulate ctx with WMMA.
template <bool PASS1>
__global__ __launch_bounds__(32 * WPB) void FractalAttention_main(
    const _Float16* __restrict__ xh, const _Float16* __restrict__ xhT,
    const float* __restrict__ xn, const float* __restrict__ gamma,
    float* __restrict__ ctx, float* __restrict__ attn, float* __restrict__ invrow)
{
  __shared__ __align__(16) _Float16 P1[32 * 64];        // column panel, row-major
  __shared__ __align__(16) _Float16 P2[64 * 32];        // column panel, transposed
  __shared__ __align__(16) _Float16 Estage[WPB][16 * 32]; // per-wave E transpose buf

  const int tid  = threadIdx.x;
  const int wave = tid >> 5;
  const int lane = tid & 31;
  const int ln   = lane & 15;
  const int hi8  = (lane < 16) ? 0 : 8;     // row-group offset (C/D layout)
  const int hi16 = (lane < 16) ? 0 : 16;    // k-group offset (B layout)

  const int qt = blockIdx.x * WPB + wave;   // 16-row query tile id
  const int b  = qt / (S / 16);
  const int q0 = (qt % (S / 16)) * 16;

  const float g = gamma[0];

  // A operand: Q tile rows as f16, two K=32 blocks (A layout: lane row = ln)
  const _Float16* qrow = xh + ((size_t)b * S + q0 + ln) * D;
  const v16h QA0 = ld16h(qrow + 0  + hi16, qrow + 8 + hi16);
  const v16h QA1 = ld16h(qrow + 32 + hi16, qrow + 40 + hi16);

  // query norms for this lane's 8 rows (rows hi8..hi8+7)
  float qn[8];
  {
    const float* qp = xn + (size_t)b * S + q0 + hi8;
    v4f a = *(const v4f*)qp;
    v4f c2 = *(const v4f*)(qp + 4);
    qn[0] = a[0];  qn[1] = a[1];  qn[2] = a[2];  qn[3] = a[3];
    qn[4] = c2[0]; qn[5] = c2[1]; qn[6] = c2[2]; qn[7] = c2[3];
  }

  float inv[8];
  if (!PASS1) {
    const float* ip = invrow + (size_t)b * S + q0 + hi8;
    v4f a = *(const v4f*)ip;
    v4f c2 = *(const v4f*)(ip + 4);
    inv[0] = a[0];  inv[1] = a[1];  inv[2] = a[2];  inv[3] = a[3];
    inv[4] = c2[0]; inv[5] = c2[1]; inv[6] = c2[2]; inv[7] = c2[3];
  }

  float rs[8] = {0.f, 0.f, 0.f, 0.f, 0.f, 0.f, 0.f, 0.f};
  v8f ctxacc[4] = {};

  for (int c = 0; c < S; c += 32) {
    // ---- stage column panels into LDS ----
#if USE_TDM
    if (wave == 0) {
      // P1: 32 rows x 64 dims, row-major (stride D)
      tdm_load_2d((unsigned)(uintptr_t)&P1[0],
                  xh + ((size_t)b * S + c) * D,
                  /*tensor_d0=*/D, /*tensor_d1=*/S,
                  /*tile_d0=*/D, /*tile_d1=*/32, /*stride0=*/D);
      if (!PASS1) {
        // P2: 64 dim-rows x 32 cols from transposed copy (stride S)
        tdm_load_2d((unsigned)(uintptr_t)&P2[0],
                    xhT + (size_t)b * D * S + c,
                    /*tensor_d0=*/S, /*tensor_d1=*/D,
                    /*tile_d0=*/32, /*tile_d1=*/D, /*stride0=*/S);
      }
      __builtin_amdgcn_s_wait_tensorcnt(0);
    }
#else
    {
      const _Float16* s1 = xh + ((size_t)b * S + c) * D;
      for (int i = tid; i < (32 * 64) / 8; i += 32 * WPB)
        *(v8h*)(P1 + i * 8) = *(const v8h*)(s1 + i * 8);
      if (!PASS1) {
        const _Float16* s2 = xhT + (size_t)b * D * S + c;
        for (int i = tid; i < (64 * 32) / 8; i += 32 * WPB) {
          int r = i >> 2, sg = i & 3;
          *(v8h*)(P2 + r * 32 + sg * 8) = *(const v8h*)(s2 + (size_t)r * S + sg * 8);
        }
      }
    }
#endif
    if (tid == 64 && c + 32 < S)  // prefetch next panel into L2
      __builtin_prefetch(xh + ((size_t)b * S + c + 32) * D, 0, 0);
    __syncthreads();

    const float cn0 = xn[(size_t)b * S + c + ln];
    const float cn1 = xn[(size_t)b * S + c + 16 + ln];
    _Float16* es = &Estage[wave][0];

#pragma unroll
    for (int t = 0; t < 2; ++t) {
      // ---- Gram subtile G = Q * Xc^T, K=64 in two WMMA steps ----
      v8f G = {};
      const _Float16* prow = &P1[(16 * t + ln) * 64];
      G = wmma_f32_f16(QA0, ld16h(prow + hi16,      prow + hi16 + 8),  G);
      G = wmma_f32_f16(QA1, ld16h(prow + 32 + hi16, prow + 40 + hi16), G);

      const float cnv = t ? cn1 : cn0;
#pragma unroll
      for (int r = 0; r < 8; ++r) {
        float sq = qn[r] + cnv - 2.0f * G[r];
        sq = fmaxf(sq, 0.0f);
        float dd = fast_sqrtf(sq);                        // raw v_sqrt_f32
        // mask keyed off sq (select doesn't wait on sqrt); d>TAU <=> sq>TAU^2
        float w  = (sq <= TAU * TAU) ? __expf(-g * dd) : 0.0f;
        float e  = __expf(w);                             // softmax numerator
        if (PASS1) {
          rs[r] += e;
        } else {
          float en = e * inv[r];                          // normalized attn
          attn[((size_t)b * S + q0 + r + hi8) * S + (size_t)(c + 16 * t + ln)] = en;
          es[(r + hi8) * 32 + 16 * t + ln] = (_Float16)en; // stage for transpose
        }
      }
    }

    if (!PASS1) {
      // ---- ctx += attn_tile(16x32) * Xc(32x64), four N=16 chunks ----
      const _Float16* ep = es + ln * 32;
      const v16h EA = ld16h(ep + hi16, ep + 8 + hi16);      // A layout
#pragma unroll
      for (int dc = 0; dc < 4; ++dc) {
        const _Float16* pp = &P2[(dc * 16 + ln) * 32 + hi16];
        ctxacc[dc] = wmma_f32_f16(EA, ld16h(pp, pp + 8), ctxacc[dc]);
      }
    }
    __syncthreads();
  }

  if (PASS1) {
    // Reduce row sums within the 16-lane groups that share the same row set.
#pragma unroll
    for (int r = 0; r < 8; ++r) {
      float v = rs[r];
      v += __shfl_xor(v, 1, 32);
      v += __shfl_xor(v, 2, 32);
      v += __shfl_xor(v, 4, 32);
      v += __shfl_xor(v, 8, 32);
      rs[r] = 1.0f / v;
    }
    if (lane == 0) {
#pragma unroll
      for (int r = 0; r < 8; ++r) invrow[(size_t)b * S + q0 + r] = rs[r];
    }
    if (lane == 16) {
#pragma unroll
      for (int r = 0; r < 8; ++r) invrow[(size_t)b * S + q0 + 8 + r] = rs[r];
    }
  } else {
    // ctx accumulated from normalized attn already — store directly.
#pragma unroll
    for (int dc = 0; dc < 4; ++dc) {
#pragma unroll
      for (int r = 0; r < 8; ++r) {
        ctx[((size_t)b * S + q0 + r + hi8) * D + dc * 16 + ln] = ctxacc[dc][r];
      }
    }
  }
}

extern "C" void kernel_launch(void* const* d_in, const int* in_sizes, int n_in,
                              void* d_out, int out_size, void* d_ws, size_t ws_size,
                              hipStream_t stream) {
  (void)in_sizes; (void)n_in; (void)out_size; (void)ws_size;
  const int*   X     = (const int*)d_in[0];
  const float* emb   = (const float*)d_in[1];
  // d_in[2] = cluster_centers: computed-then-discarded in the reference; unused.
  const float* gamma = (const float*)d_in[3];

  float* out   = (float*)d_out;
  float* ctx   = out;                          // [B,S,D] f32
  float* attnp = out + (size_t)B * S * D;      // [B,S,S] f32

  _Float16* xh     = (_Float16*)d_ws;                    // [B,S,D] f16 (2 MB)
  _Float16* xhT    = xh + (size_t)B * S * D;             // [B,D,S] f16 (2 MB)
  float*    xn     = (float*)(xhT + (size_t)B * S * D);  // [B,S] row norms
  float*    invrow = xn + (size_t)B * S;                 // [B,S] 1/rowsum

  FractalAttention_embed<<<(B * S) / 8, 256, 0, stream>>>(X, emb, xh, xhT, xn);

  const int blocks = (B * S / 16) / WPB;       // 1024 query tiles / 4 waves
  FractalAttention_main<true ><<<blocks, 32 * WPB, 0, stream>>>(
      xh, xhT, xn, gamma, ctx, attnp, invrow);
  FractalAttention_main<false><<<blocks, 32 * WPB, 0, stream>>>(
      xh, xhT, xn, gamma, ctx, attnp, invrow);
}
